// GraphSAGENet_79980880986661
// MI455X (gfx1250) — matrix operand order repaired and checked by reading, not compiled, not run on hardware
//
#include <hip/hip_runtime.h>
#include <hip/hip_bf16.h>
#include <math.h>

#define NNODES 100000
#define NEDGES 1600000
#define NC     128
#define NLAYERS 7
#define SLOPE  0.02f

typedef __attribute__((ext_vector_type(16))) __bf16 v16bf;
typedef __attribute__((ext_vector_type(8)))  float  v8f;

#define WFRAG (4 * 8 * 32 * 16)   // bf16 elements per packed weight block (32 KB)

// ---- order-preserving float<->uint encoding for atomic max ----------------
// enc is strictly monotonic over floats; sentinel 0u is below enc(-inf),
// so aggE==0 after reduction  <=>  node had no in-edges  =>  0.0 (PyG conv).
__device__ __forceinline__ unsigned encf(float f) {
    unsigned u = __float_as_uint(f);
    return (u & 0x80000000u) ? ~u : (u | 0x80000000u);
}
__device__ __forceinline__ float decf(unsigned u) {
    return (u & 0x80000000u) ? __uint_as_float(u & 0x7FFFFFFFu)
                             : __uint_as_float(~u);
}
__device__ __forceinline__ float dec0(unsigned u) { return u ? decf(u) : 0.0f; }

// ---- init agg buffer to sentinel ------------------------------------------
__global__ void fill_zero_kernel(unsigned* __restrict__ p, int n) {
    int i = blockIdx.x * blockDim.x + threadIdx.x;
    if (i < n) p[i] = 0u;
}

// ---- edge scatter-max: agg[dst] = max(agg[dst], h[src]) -------------------
// 8 threads per edge, 16 channels each; float4 gathers; h is L2-resident.
__global__ void scatter_max_kernel(const float* __restrict__ h,
                                   const int* __restrict__ src,
                                   const int* __restrict__ dst,
                                   unsigned* __restrict__ aggE) {
    unsigned gid = blockIdx.x * blockDim.x + threadIdx.x;
    unsigned e = gid >> 3;
    if (e >= NEDGES) return;
    int part = gid & 7;
    int s = src[e], d = dst[e];
    const float4* hp = (const float4*)(h + (size_t)s * NC + part * 16);
    unsigned* ap = aggE + (size_t)d * NC + part * 16;
#pragma unroll
    for (int q = 0; q < 4; q++) {
        float4 v = hp[q];
        atomicMax(ap + q * 4 + 0, encf(v.x));
        atomicMax(ap + q * 4 + 1, encf(v.y));
        atomicMax(ap + q * 4 + 2, encf(v.z));
        atomicMax(ap + q * 4 + 3, encf(v.w));
    }
}

// ---- one-time weight pack: W[o][k] (f32) -> bf16 B-fragments --------------
// B = W^T (K x N). Per (chunk c, n-tile t, lane): lane holds column
// n = t*16 + (lane&15), k = c*32 + (lane>>4)*8 + (i < 8 ? i : i + 8)  (i = 0..15).
__global__ void pack_weights_kernel(const float* __restrict__ Wl,
                                    const float* __restrict__ Wr,
                                    __bf16* __restrict__ wpack) {
    int gid = blockIdx.x * blockDim.x + threadIdx.x;
    if (gid >= NLAYERS * 2 * 4 * 8 * 32) return;
    int lane  = gid & 31;
    int t     = (gid >> 5) & 7;
    int c     = (gid >> 8) & 3;
    int which = (gid >> 10) & 1;
    int layer = gid >> 11;
    const float* W = (which ? Wr : Wl) + (size_t)layer * NC * NC;
    int n  = t * 16 + (lane & 15);
    int k0 = c * 32 + (lane >> 4) * 8;
    __bf16* op = wpack + ((((size_t)(layer * 2 + which) * 4 + c) * 8 + t) * 32 + lane) * 16;
#pragma unroll
    for (int i = 0; i < 16; i++) op[i] = (__bf16)W[n * NC + k0 + (i < 8 ? i : i + 8)];
}

// ---- fused SAGE layer: h = leaky(agg @ Wl^T + bl + h @ Wr^T)  (in place) --
// 256 threads = 8 waves; wave w owns output rows [blk*128 + w*16, +16) x 128
// cols. Both weight blocks (64 KB) are staged in LDS once per workgroup (all
// 8 waves consume identical lane-indexed B fragments), so the K-loop's B
// reads are ds_load_b128 with immediate offsets and vmem carries only the
// two A-operand streams. 64 v_wmma_f32_16x16x32_bf16 per wave. In-place is
// safe: each wave reads only the rows it writes (tail lanes clamp-read,
// never store).
__global__ __launch_bounds__(256) void sage_gemm_kernel(
        const unsigned* __restrict__ aggE,
        float* __restrict__ h,
        const __bf16* __restrict__ wl,
        const __bf16* __restrict__ wr,
        const float* __restrict__ bias) {
    __shared__ __align__(16) __bf16 lwl[WFRAG];   // 32 KB
    __shared__ __align__(16) __bf16 lwr[WFRAG];   // 32 KB

    {   // cooperative 64 KB stage: 256 threads x 8 float4 each per block
        const float4* gl = (const float4*)wl;
        const float4* gr = (const float4*)wr;
        float4* sl = (float4*)lwl;
        float4* sr = (float4*)lwr;
#pragma unroll
        for (int i = 0; i < WFRAG / 8 / 256; i++) {
            int idx = i * 256 + threadIdx.x;
            sl[idx] = gl[idx];
            sr[idx] = gr[idx];
        }
    }
    __syncthreads();

    int lane = threadIdx.x & 31;
    int wave = threadIdx.x >> 5;
    int row_base = blockIdx.x * 128 + wave * 16;
    int m    = lane & 15;
    int half = lane >> 4;
    int row  = row_base + m;
    int rowc = row < NNODES ? row : NNODES - 1;
    const float*    hrow = h    + (size_t)rowc * NC;
    const unsigned* arow = aggE + (size_t)rowc * NC;

    v8f acc[8];
#pragma unroll
    for (int t = 0; t < 8; t++) {
        float b = bias[t * 16 + m];          // bias per output column n
#pragma unroll
        for (int i = 0; i < 8; i++) acc[t][i] = b;
    }

#pragma unroll
    for (int c = 0; c < 4; c++) {            // K = 128 in chunks of 32
        int kA = c * 32 + half * 8;          // ISA A-layout: K halves per lane
        float4 h0 = *(const float4*)(hrow + kA);
        float4 h1 = *(const float4*)(hrow + kA + 4);
        float4 h2 = *(const float4*)(hrow + kA + 16);
        float4 h3 = *(const float4*)(hrow + kA + 20);
        uint4  a0 = *(const uint4*)(arow + kA);
        uint4  a1 = *(const uint4*)(arow + kA + 4);
        uint4  a2 = *(const uint4*)(arow + kA + 16);
        uint4  a3 = *(const uint4*)(arow + kA + 20);

        v16bf aH, aA;
        aH[0] = (__bf16)h0.x; aH[1] = (__bf16)h0.y; aH[2]  = (__bf16)h0.z; aH[3]  = (__bf16)h0.w;
        aH[4] = (__bf16)h1.x; aH[5] = (__bf16)h1.y; aH[6]  = (__bf16)h1.z; aH[7]  = (__bf16)h1.w;
        aH[8] = (__bf16)h2.x; aH[9] = (__bf16)h2.y; aH[10] = (__bf16)h2.z; aH[11] = (__bf16)h2.w;
        aH[12]= (__bf16)h3.x; aH[13]= (__bf16)h3.y; aH[14] = (__bf16)h3.z; aH[15] = (__bf16)h3.w;
        aA[0] = (__bf16)dec0(a0.x); aA[1]  = (__bf16)dec0(a0.y);
        aA[2] = (__bf16)dec0(a0.z); aA[3]  = (__bf16)dec0(a0.w);
        aA[4] = (__bf16)dec0(a1.x); aA[5]  = (__bf16)dec0(a1.y);
        aA[6] = (__bf16)dec0(a1.z); aA[7]  = (__bf16)dec0(a1.w);
        aA[8] = (__bf16)dec0(a2.x); aA[9]  = (__bf16)dec0(a2.y);
        aA[10]= (__bf16)dec0(a2.z); aA[11] = (__bf16)dec0(a2.w);
        aA[12]= (__bf16)dec0(a3.x); aA[13] = (__bf16)dec0(a3.y);
        aA[14]= (__bf16)dec0(a3.z); aA[15] = (__bf16)dec0(a3.w);

#pragma unroll
        for (int t = 0; t < 8; t++) {        // 8 n-tiles of 16 cols
            v16bf bL = *(const v16bf*)(lwl + (((size_t)c * 8 + t) * 32 + lane) * 16);
            v16bf bR = *(const v16bf*)(lwr + (((size_t)c * 8 + t) * 32 + lane) * 16);
            acc[t] = __builtin_amdgcn_wmma_f32_16x16x32_bf16(
                         false, aA, false, bL, (short)0, acc[t], false, false);
            acc[t] = __builtin_amdgcn_wmma_f32_16x16x32_bf16(
                         false, aH, false, bR, (short)0, acc[t], false, false);
        }
    }

    // C/D layout: lane -> col n = t*16 + m; VGPR v -> row v + 8*half
#pragma unroll
    for (int t = 0; t < 8; t++) {
#pragma unroll
        for (int v = 0; v < 8; v++) {
            int r = row_base + v + half * 8;
            if (r < NNODES) {
                float x = acc[t][v];
                h[(size_t)r * NC + t * 16 + m] = (x >= 0.f) ? x : SLOPE * x;
            }
        }
    }
}

// ---- output layer: 3 channels + tanh*0.5, one wave per node ---------------
__global__ void out_kernel(const unsigned* __restrict__ aggE,
                           const float* __restrict__ h,
                           const float* __restrict__ WlO,
                           const float* __restrict__ blO,
                           const float* __restrict__ WrO,
                           float* __restrict__ out) {
    int gid  = blockIdx.x * blockDim.x + threadIdx.x;
    int node = gid >> 5;
    int lane = gid & 31;
    if (node >= NNODES) return;
    const float*    hr = h    + (size_t)node * NC;
    const unsigned* ar = aggE + (size_t)node * NC;
    float av[4], hv[4];
#pragma unroll
    for (int q = 0; q < 4; q++) {
        int k = lane + 32 * q;
        av[q] = dec0(ar[k]);
        hv[q] = hr[k];
    }
#pragma unroll
    for (int o = 0; o < 3; o++) {
        float s = 0.f;
#pragma unroll
        for (int q = 0; q < 4; q++) {
            int k = lane + 32 * q;
            s = fmaf(av[q], WlO[o * NC + k], s);
            s = fmaf(hv[q], WrO[o * NC + k], s);
        }
#pragma unroll
        for (int off = 16; off > 0; off >>= 1) s += __shfl_xor(s, off, 32);
        if (lane == 0) out[node * 3 + o] = tanhf(s + blO[o]) * 0.5f;
    }
}

extern "C" void kernel_launch(void* const* d_in, const int* in_sizes, int n_in,
                              void* d_out, int out_size, void* d_ws, size_t ws_size,
                              hipStream_t stream) {
    const float* x    = (const float*)d_in[0];
    const int*   eidx = (const int*)  d_in[1];
    const float* Wl   = (const float*)d_in[2];
    const float* bl   = (const float*)d_in[3];
    const float* Wr   = (const float*)d_in[4];
    const float* WlO  = (const float*)d_in[5];
    const float* blO  = (const float*)d_in[6];
    const float* WrO  = (const float*)d_in[7];
    float* out = (float*)d_out;

    char* ws = (char*)d_ws;
    size_t hbytes = (size_t)NNODES * NC * sizeof(float);          // 51.2 MB
    float*    h     = (float*)ws;                                 // activations (in-place)
    unsigned* aggE  = (unsigned*)(ws + hbytes);                   // encoded max-agg
    __bf16*   wpack = (__bf16*)(ws + 2 * hbytes);                 // packed bf16 weights

    const int* src = eidx;
    const int* dst = eidx + NEDGES;

    hipMemcpyAsync(h, x, hbytes, hipMemcpyDeviceToDevice, stream);
    pack_weights_kernel<<<(NLAYERS * 2 * 4 * 8 * 32 + 255) / 256, 256, 0, stream>>>(
        Wl, Wr, wpack);

    const int fillBlocks = (NNODES * NC + 255) / 256;
    const int scatBlocks = (NEDGES * 8 + 255) / 256;
    const int gemmBlocks = (NNODES + 127) / 128;

    for (int layer = 0; layer < NLAYERS; ++layer) {
        fill_zero_kernel<<<fillBlocks, 256, 0, stream>>>(aggE, NNODES * NC);
        scatter_max_kernel<<<scatBlocks, 256, 0, stream>>>(h, src, dst, aggE);
        sage_gemm_kernel<<<gemmBlocks, 256, 0, stream>>>(
            aggE, h,
            wpack + ((size_t)layer * 2 + 0) * WFRAG,
            wpack + ((size_t)layer * 2 + 1) * WFRAG,
            bl + (size_t)layer * NC);
    }

    // final aggregation over h7 (already leaky'd, matching wrapper), output conv
    fill_zero_kernel<<<fillBlocks, 256, 0, stream>>>(aggE, NNODES * NC);
    scatter_max_kernel<<<scatBlocks, 256, 0, stream>>>(h, src, dst, aggE);
    out_kernel<<<(NNODES * 32 + 255) / 256, 256, 0, stream>>>(aggE, h, WlO, blO, WrO, out);
}
